// BezierButtress_9500467659454
// MI455X (gfx1250) — compile-verified, matrix operand order checked by reading, not run on hardware
//
#include <hip/hip_runtime.h>
#include <hip/hip_bf16.h>

typedef __attribute__((ext_vector_type(16))) __bf16 bf16x16;
typedef __attribute__((ext_vector_type(8)))  float  f32x8;

#define ORDER 16
#define KDIM  17
#define DDIM  32
#define PDIM  20
#define NPTS  24576
#define NSTEP 31      // D-1 chained matmul steps
#define WTILE 512     // 32 lanes * 16 bf16 per packed A tile
#define WAVES 4       // waves per block (independent tiles)

// ---------------------------------------------------------------------------
// Prep: zero-padded, per-lane-packed bf16 W^T operands (A-operand layout).
// wpk[(((pass*P + p)*31 + dd)*2 + jtile)*512 + lane*16 + slot]
//   row  M = jtile*16 + (lane&15)  -> output index j
//   slot s -> K: k = s + 8*(lane>=16) + (s>=8 ? 8 : 0)
//   value = A[M][K] = W^T[j][k] = W[k][j];  zero when k>=17 or j>=17.
// Var pass folds in exp(varw)*prior_sc[j]^2.
// ---------------------------------------------------------------------------
__global__ void prep_weights(const float* __restrict__ meanw,
                             const float* __restrict__ varw,
                             const float* __restrict__ prior_sc,
                             __bf16* __restrict__ wpk) {
  int idx = blockIdx.x * blockDim.x + threadIdx.x;
  const int total = 2 * PDIM * NSTEP * 2 * WTILE;
  if (idx >= total) return;
  int slot = idx & 15;
  int lane = (idx >> 4) & 31;
  int rest = idx >> 9;
  int jt = rest & 1;            rest >>= 1;
  int dd = rest % NSTEP;        rest /= NSTEP;
  int p  = rest % PDIM;
  int pass = rest / PDIM;
  int k = slot + ((lane & 16) ? 8 : 0) + ((slot >= 8) ? 8 : 0);
  int j = jt * 16 + (lane & 15);
  float v = 0.f;
  if (k < KDIM && j < KDIM) {
    int off = ((dd * PDIM + p) * KDIM + k) * KDIM + j;
    if (pass == 0) {
      v = meanw[off];
    } else {
      float sc = prior_sc[j];
      v = __expf(varw[off]) * sc * sc;
    }
  }
  wpk[idx] = (__bf16)v;
}

// ---------------------------------------------------------------------------
// Per-lane Bernstein factors for both passes: lane owns point n and
// j = 8*grp .. 8*grp+7 (plus j=16).  bm = b, bv = b^2.
// ---------------------------------------------------------------------------
__device__ __forceinline__ void bern_lane2(float x, int grp,
                                           float bm[8], float& b16m,
                                           float bv[8], float& b16v) {
  constexpr float BIN[17] = {1.f, 16.f, 120.f, 560.f, 1820.f, 4368.f, 8008.f,
                             11440.f, 12870.f, 11440.f, 8008.f, 4368.f,
                             1820.f, 560.f, 120.f, 16.f, 1.f};
  float s = 1.f - x;
  float x2 = x * x, x4 = x2 * x2, x8 = x4 * x4;
  float s2 = s * s, s4 = s2 * s2, s8 = s4 * s4;
  float xpow[8];
  xpow[0] = grp ? x8 : 1.f;
#pragma unroll
  for (int i = 1; i < 8; ++i) xpow[i] = xpow[i - 1] * x;
  float se = grp ? s : s8 * s;  // s^(16-(j0+7)): grp0 -> s^9, grp1 -> s^1
  bm[7] = (grp ? BIN[15] : BIN[7]) * xpow[7] * se;
#pragma unroll
  for (int i = 6; i >= 0; --i) {
    se *= s;
    bm[i] = (grp ? BIN[8 + i] : BIN[i]) * xpow[i] * se;
  }
  b16m = x8 * x8;  // C(16,16)=1
#pragma unroll
  for (int i = 0; i < 8; ++i) bv[i] = bm[i] * bm[i];
  b16v = b16m * b16m;
}

// Re-pack D accumulators (lane=n, j striped) into next-step B operand:
// slots 0..7 = j0..j0+7; slot 8 = j=16 (grp0 only, else K=24 -> 0); 9..15 = 0.
__device__ __forceinline__ bf16x16 pack_b(const f32x8& c0, const f32x8& c1,
                                          int grp) {
  bf16x16 b;
#pragma unroll
  for (int e = 0; e < 8; ++e) b[e] = (__bf16)c0[e];
  b[8] = (__bf16)(grp == 0 ? c1[0] : 0.f);
#pragma unroll
  for (int e = 9; e < 16; ++e) b[e] = (__bf16)0.f;
  return b;
}

// ---------------------------------------------------------------------------
// Main: one wave per 16-point tile, BOTH passes fused (two independent WMMA
// chains interleaved to fill the D->B dependency latency). Register-resident:
// D^T = (W^T) x (f^T), 4x v_wmma_f32_16x16x32_bf16 per step, no LDS in the
// steady state beyond the x broadcast.
// ---------------------------------------------------------------------------
__global__ __launch_bounds__(32 * WAVES) void bezier_main(
    const float* __restrict__ X,         // N*D
    const float* __restrict__ meanw0,    // P*K
    const float* __restrict__ varw0,     // P*K
    const float* __restrict__ prior_sc,  // K
    const float* __restrict__ post_prec, // P
    const int*   __restrict__ perm,      // P*D
    const __bf16* __restrict__ wpk,      // packed W^T tiles in ws
    float* __restrict__ out)             // [mean N | var N]
{
  __shared__ float xt[WAVES][16 * DDIM];  // 16-point X tile per wave

  const int wv   = threadIdx.x >> 5;
  const int lane = threadIdx.x & 31;
  const int col  = lane & 15;             // point index within tile
  const int grp  = lane >> 4;             // j-half: 0 -> j 0..7, 1 -> 8..15
  const int tile = blockIdx.x * WAVES + wv;
  const int n0   = tile * 16;

  // Stage the X tile (512 floats) once: 4 x b128 per lane.
  {
    const float4* src = (const float4*)(X + (size_t)n0 * DDIM);
    float4* dst = (float4*)xt[wv];
#pragma unroll
    for (int i = 0; i < 4; ++i) dst[lane + 32 * i] = src[lane + 32 * i];
  }
  __syncthreads();

  const float* myx = xt[wv] + col * DDIM;
  float accm = 0.f, accv = 0.f;

  for (int p = 0; p < PDIM; ++p) {
    // ---- d = 0: f0[n][j] = w0[j] * b0[n][j], directly in D layout ----
    float bm[8], bvv[8], b16m, b16v;
    bern_lane2(myx[perm[p * DDIM + 0]], grp, bm, b16m, bvv, b16v);

    f32x8 c0m, c1m = {}, c0v, c1v = {};
    {
      const int j0 = 8 * grp;
#pragma unroll
      for (int v = 0; v < 8; ++v) {
        c0m[v] = meanw0[p * KDIM + j0 + v] * bm[v];
        float sc = prior_sc[j0 + v];
        c0v[v] = __expf(varw0[p * KDIM + j0 + v]) * sc * sc * bvv[v];
      }
      if (grp == 0) {
        c1m[0] = meanw0[p * KDIM + 16] * b16m;
        float sc = prior_sc[16];
        c1v[0] = __expf(varw0[p * KDIM + 16]) * sc * sc * b16v;
      }
    }
    bf16x16 bvM = pack_b(c0m, c1m, grp);
    bf16x16 bvV = pack_b(c0v, c1v, grp);

    // ---- d = 1..31: 4 WMMAs per step, two independent chains ----
    const __bf16* wbM = wpk + ((size_t)p * NSTEP) * 2 * WTILE;
    const __bf16* wbV = wpk + ((size_t)(PDIM + p) * NSTEP) * 2 * WTILE;
    for (int d = 1; d < DDIM; ++d, wbM += 2 * WTILE, wbV += 2 * WTILE) {
      bf16x16 a0m = *(const bf16x16*)(wbM + lane * 16);
      bf16x16 a1m = *(const bf16x16*)(wbM + WTILE + lane * 16);
      bf16x16 a0v = *(const bf16x16*)(wbV + lane * 16);
      bf16x16 a1v = *(const bf16x16*)(wbV + WTILE + lane * 16);
      if (d + 1 < DDIM) {
        __builtin_prefetch(wbM + 2 * WTILE + lane * 16, 0, 3);
        __builtin_prefetch(wbV + 2 * WTILE + lane * 16, 0, 3);
      }

      bern_lane2(myx[perm[p * DDIM + d]], grp, bm, b16m, bvv, b16v);

      f32x8 z = {};
      c0m = __builtin_amdgcn_wmma_f32_16x16x32_bf16(false, a0m, false, bvM,
                                                    (short)0, z, false, false);
      c0v = __builtin_amdgcn_wmma_f32_16x16x32_bf16(false, a0v, false, bvV,
                                                    (short)0, z, false, false);
      c1m = __builtin_amdgcn_wmma_f32_16x16x32_bf16(false, a1m, false, bvM,
                                                    (short)0, z, false, false);
      c1v = __builtin_amdgcn_wmma_f32_16x16x32_bf16(false, a1v, false, bvV,
                                                    (short)0, z, false, false);
#pragma unroll
      for (int v = 0; v < 8; ++v) {
        c0m[v] *= bm[v];     // elementwise basis (mean chain)
        c0v[v] *= bvv[v];    // elementwise basis^2 (var chain)
      }
      c1m[0] *= b16m;        // j=16 (structurally zero for grp1)
      c1v[0] *= b16v;
      bvM = pack_b(c0m, c1m, grp);
      bvV = pack_b(c0v, c1v, grp);
    }

    // ---- reduce over j for this p ----
    float sm = 0.f, sv = 0.f;
#pragma unroll
    for (int v = 0; v < 8; ++v) { sm += c0m[v]; sv += c0v[v]; }
    if (grp == 0) { sm += c1m[0]; sv += c1v[0]; }
    accm += sm + __shfl_xor(sm, 16, 32);
    float totv = sv + __shfl_xor(sv, 16, 32);
    accv += totv / post_prec[p];
  }

  if (lane < 16) {
    out[n0 + col]        = accm;
    out[NPTS + n0 + col] = accv;
  }
}

// ---------------------------------------------------------------------------
extern "C" void kernel_launch(void* const* d_in, const int* in_sizes, int n_in,
                              void* d_out, int out_size, void* d_ws, size_t ws_size,
                              hipStream_t stream) {
  const float* X         = (const float*)d_in[0];
  const float* meanw0    = (const float*)d_in[1];
  const float* meanw     = (const float*)d_in[2];
  const float* varw0     = (const float*)d_in[3];
  const float* varw      = (const float*)d_in[4];
  const float* prior_sc  = (const float*)d_in[5];
  const float* post_prec = (const float*)d_in[6];
  const int*   perm      = (const int*)d_in[7];
  __bf16* wpk = (__bf16*)d_ws;
  float* out  = (float*)d_out;

  const int total = 2 * PDIM * NSTEP * 2 * WTILE;
  prep_weights<<<(total + 255) / 256, 256, 0, stream>>>(meanw, varw, prior_sc, wpk);

  dim3 grid(NPTS / (16 * WAVES));
  bezier_main<<<grid, 32 * WAVES, 0, stream>>>(X, meanw0, varw0, prior_sc,
                                               post_prec, perm, wpk, out);
}
